// JetGNN_46256797778449
// MI455X (gfx1250) — matrix operand order repaired and checked by reading, not compiled, not run on hardware
//
#include <hip/hip_runtime.h>
#include <hip/hip_bf16.h>

#define NNODES 100000
#define NEDGES 3200000
#define NGRAPH 512

typedef __attribute__((ext_vector_type(16))) _Float16 v16h;
typedef __attribute__((ext_vector_type(8)))  float    v8f;

// ---------------- degree + 6-dim input scatter ----------------
__global__ __launch_bounds__(256) void deg_scatter6_kernel(
    const int* __restrict__ ei, const float* __restrict__ x,
    float* __restrict__ accum6, float* __restrict__ deg, int nEdges)
{
    int e = blockIdx.x * 256 + threadIdx.x;
    if (e >= nEdges) return;
    int src = ei[e];
    int dst = ei[nEdges + e];
    atomicAdd(&deg[dst], 1.0f);
#pragma unroll
    for (int k = 0; k < 6; ++k)
        atomicAdd(&accum6[dst * 6 + k], x[src * 6 + k]);
}

__global__ __launch_bounds__(256) void invert_deg_kernel(float* deg, int n)
{
    int i = blockIdx.x * 256 + threadIdx.x;
    if (i < n) deg[i] = 1.0f / fmaxf(deg[i], 1.0f);
}

// ---------------- layer 1 (K=6, scalar) ----------------
__global__ __launch_bounds__(256) void sage1_kernel(
    const float* __restrict__ x, const float* __restrict__ accum6,
    const float* __restrict__ deg_inv,
    const float* __restrict__ Wl, const float* __restrict__ bl,
    const float* __restrict__ Wr, float* __restrict__ out, int nNodes)
{
    __shared__ float wl[6 * 64], wr[6 * 64], bb[64];
    for (int i = threadIdx.x; i < 384; i += 256) { wl[i] = Wl[i]; wr[i] = Wr[i]; }
    if (threadIdx.x < 64) bb[threadIdx.x] = bl[threadIdx.x];
    __syncthreads();
    int idx = blockIdx.x * 256 + threadIdx.x;
    if (idx >= nNodes * 64) return;
    int n = idx >> 6, j = idx & 63;
    float inv = deg_inv[n];
    float acc = bb[j];
#pragma unroll
    for (int k = 0; k < 6; ++k)
        acc += accum6[n * 6 + k] * inv * wl[k * 64 + j] + x[n * 6 + k] * wr[k * 64 + j];
    out[idx] = acc;
}

// ---------------- BatchNorm statistics ----------------
__global__ __launch_bounds__(256) void colstats_kernel(
    const float* __restrict__ h, float* __restrict__ sums,
    float* __restrict__ sumsq, int nNodes)
{
    __shared__ float ls[256], lq[256];
    int f = threadIdx.x & 63;
    int sub = threadIdx.x >> 6; // 0..3
    int n0 = blockIdx.x * 256;
    float s = 0.f, q = 0.f;
    for (int i = sub; i < 256; i += 4) {
        int n = n0 + i;
        if (n < nNodes) { float v = h[n * 64 + f]; s += v; q += v * v; }
    }
    ls[threadIdx.x] = s; lq[threadIdx.x] = q;
    __syncthreads();
    if (sub == 0) {
#pragma unroll
        for (int j = 1; j < 4; ++j) { s += ls[f + j * 64]; q += lq[f + j * 64]; }
        atomicAdd(&sums[f], s);
        atomicAdd(&sumsq[f], q);
    }
}

__global__ __launch_bounds__(256) void bn_relu_kernel(
    float* __restrict__ h, const float* __restrict__ sums,
    const float* __restrict__ sumsq, const float* __restrict__ g,
    const float* __restrict__ be, int nNodes)
{
    int idx = blockIdx.x * 256 + threadIdx.x;
    if (idx >= nNodes * 64) return;
    int f = idx & 63;
    float invN = 1.0f / (float)nNodes;
    float mu = sums[f] * invN;
    float var = sumsq[f] * invN - mu * mu;
    float v = (h[idx] - mu) * rsqrtf(var + 1e-5f) * g[f] + be[f];
    h[idx] = fmaxf(v, 0.0f);
}

// ---------------- 64-dim feature scatter (layers 2/3) ----------------
__global__ __launch_bounds__(256) void scatter64_kernel(
    const int* __restrict__ ei, const float* __restrict__ h,
    float* __restrict__ accum, int nEdges)
{
    int gid = blockIdx.x * 256 + threadIdx.x; // nEdges*16 = 51.2M < 2^31
    if (gid >= nEdges * 16) return;
    int e = gid >> 4, q = gid & 15;
    int src = ei[e];
    int dst = ei[nEdges + e];
    float4 v = ((const float4*)h)[src * 16 + q];
    float* base = &accum[dst * 64 + q * 4];
    atomicAdd(base + 0, v.x);
    atomicAdd(base + 1, v.y);
    atomicAdd(base + 2, v.z);
    atomicAdd(base + 3, v.w);
}

// ---------------- f16 weight prep: W = [Wl ; Wr] (128x64) ----------------
__global__ __launch_bounds__(256) void prep_w16_kernel(
    const float* __restrict__ Wl, const float* __restrict__ Wr,
    _Float16* __restrict__ W16)
{
    int i = blockIdx.x * 256 + threadIdx.x; // 8192
    if (i >= 8192) return;
    int k = i >> 6, j = i & 63;
    float v = (k < 64) ? Wl[k * 64 + j] : Wr[(k - 64) * 64 + j];
    W16[i] = (_Float16)v;
}

// ---------------- fused SAGE transform via WMMA ----------------
// out[N,64] = ([accum*deg_inv | h])[N,128] @ W16[128,64] + bias
// 8 waves/block, one wave -> 16 nodes x 64 features, K=128 in 4 WMMA steps.
__global__ __launch_bounds__(256) void sage_gemm_wmma_kernel(
    const float* __restrict__ accum, const float* __restrict__ h,
    const float* __restrict__ deg_inv, const _Float16* __restrict__ W16,
    const float* __restrict__ bias, float* __restrict__ out, int nNodes)
{
    // B fragments pre-swizzled: entry ((kstep*4+ntile)*32 + lane) holds 16 halves:
    //   N = ntile*16 + (lane&15), K = kstep*32 + (lane>=16 ? 16 : 0) + j  (j=0..15)
    __shared__ _Float16 wlds[128 * 64];
    const int tid = threadIdx.x;
    for (int e = tid; e < 512; e += 256) {
        int lane16 = e & 31;
        int ntile  = (e >> 5) & 3;
        int kstep  = e >> 7;
        int ncol   = ntile * 16 + (lane16 & 15);
        int kbase  = kstep * 32 + ((lane16 >> 4) ? 16 : 0);
        _Float16* dstp = &wlds[e * 16];
#pragma unroll
        for (int j = 0; j < 16; ++j)
            dstp[j] = W16[(kbase + j) * 64 + ncol];
    }
    __syncthreads();

    const int lane  = tid & 31;
    const int wave  = tid >> 5;
    const int rowA  = lane & 15;   // M for A-fragment loading
    const int khalf = lane >> 4;
    const int tileBase = blockIdx.x * 128 + wave * 16;
    int node  = tileBase + rowA;
    int nodeC = node < nNodes ? node : (nNodes - 1); // clamp loads, mask stores
    const float inv = deg_inv[nodeC];
    const float4* accv = (const float4*)accum;
    const float4* hv   = (const float4*)h;

    v8f acc[4] = {};

#pragma unroll
    for (int kstep = 0; kstep < 4; ++kstep) {
        // A fragment (contiguous k order, same as B in LDS):
        //   halves 0..7  <- K = kstep*32 + khalf*16 + (0..7)
        //   halves 8..15 <- K = kstep*32 + khalf*16 + 8 + (0..7)
        v16h a;
#pragma unroll
        for (int r = 0; r < 2; ++r) {
            int K = kstep * 32 + khalf * 16 + r * 8;
            float4 p, q;
            if (K < 64) {                       // mean part: accum * deg_inv
                p = accv[nodeC * 16 + (K >> 2)];
                q = accv[nodeC * 16 + (K >> 2) + 1];
                p.x *= inv; p.y *= inv; p.z *= inv; p.w *= inv;
                q.x *= inv; q.y *= inv; q.z *= inv; q.w *= inv;
            } else {                            // self part: h
                int Kh = K - 64;
                p = hv[nodeC * 16 + (Kh >> 2)];
                q = hv[nodeC * 16 + (Kh >> 2) + 1];
            }
            a[r * 8 + 0] = (_Float16)p.x; a[r * 8 + 1] = (_Float16)p.y;
            a[r * 8 + 2] = (_Float16)p.z; a[r * 8 + 3] = (_Float16)p.w;
            a[r * 8 + 4] = (_Float16)q.x; a[r * 8 + 5] = (_Float16)q.y;
            a[r * 8 + 6] = (_Float16)q.z; a[r * 8 + 7] = (_Float16)q.w;
        }
#pragma unroll
        for (int ntile = 0; ntile < 4; ++ntile) {
            v16h b = *(const v16h*)&wlds[((kstep * 4 + ntile) * 32 + lane) * 16];
            acc[ntile] = __builtin_amdgcn_wmma_f32_16x16x32_f16(
                false, a, false, b, (short)0, acc[ntile], false, false);
        }
    }

    // C/D layout: VGPR r -> M = (lane>=16 ? 8 : 0) + r, N = lane&15
    const int ncol0 = lane & 15;
    const int mbase = khalf * 8;
#pragma unroll
    for (int ntile = 0; ntile < 4; ++ntile) {
        int n = ntile * 16 + ncol0;
        float bv = bias[n];
#pragma unroll
        for (int r = 0; r < 8; ++r) {
            int m = tileBase + mbase + r;
            if (m < nNodes) out[m * 64 + n] = acc[ntile][r] + bv;
        }
    }
}

// ---------------- pooling ----------------
__global__ __launch_bounds__(256) void pool_kernel(
    const float* __restrict__ h, const int* __restrict__ batch,
    float* __restrict__ pmax, float* __restrict__ psum,
    float* __restrict__ pcnt, int nNodes)
{
    int idx = blockIdx.x * 256 + threadIdx.x;
    if (idx >= nNodes * 64) return;
    int n = idx >> 6, f = idx & 63;
    int g = batch[n];
    float v = h[idx];  // v >= 0 post-ReLU: int-bit max == float max
    atomicMax((int*)&pmax[g * 64 + f], __float_as_int(v));
    atomicAdd(&psum[g * 64 + f], v);
    if (f == 0) atomicAdd(&pcnt[g], 1.0f);
}

// ---------------- MLP head ----------------
__global__ __launch_bounds__(256) void mlp1_kernel(
    const float* __restrict__ pmax, const float* __restrict__ psum,
    const float* __restrict__ pcnt, const float* __restrict__ W1,
    const float* __restrict__ b1, float* __restrict__ z)
{
    int idx = blockIdx.x * 256 + threadIdx.x; // 512*64
    if (idx >= NGRAPH * 64) return;
    int g = idx >> 6, j = idx & 63;
    float invc = 1.0f / fmaxf(pcnt[g], 1.0f);
    float acc = b1[j];
#pragma unroll 4
    for (int k = 0; k < 64; ++k) acc += pmax[g * 64 + k] * W1[k * 64 + j];
#pragma unroll 4
    for (int k = 0; k < 64; ++k) acc += psum[g * 64 + k] * invc * W1[(64 + k) * 64 + j];
    z[idx] = fmaxf(acc, 0.0f);
}

__global__ __launch_bounds__(256) void mlp2_kernel(
    const float* __restrict__ z, const float* __restrict__ W2,
    const float* __restrict__ b2, float* __restrict__ out)
{
    int idx = blockIdx.x * 256 + threadIdx.x; // 1024
    if (idx >= NGRAPH * 2) return;
    int g = idx >> 1, c = idx & 1;
    float acc = b2[c];
#pragma unroll 4
    for (int j = 0; j < 64; ++j) acc += z[g * 64 + j] * W2[j * 2 + c];
    out[idx] = acc;
}

// ---------------- host launch ----------------
extern "C" void kernel_launch(void* const* d_in, const int* in_sizes, int n_in,
                              void* d_out, int out_size, void* d_ws, size_t ws_size,
                              hipStream_t stream) {
    (void)in_sizes; (void)n_in; (void)out_size; (void)ws_size;
    const float* x      = (const float*)d_in[0];
    const int*   ei     = (const int*)d_in[1];
    const int*   batch  = (const int*)d_in[2];
    const float* Wl1 = (const float*)d_in[3];
    const float* bl1 = (const float*)d_in[4];
    const float* Wr1 = (const float*)d_in[5];
    const float* Wl2 = (const float*)d_in[6];
    const float* bl2 = (const float*)d_in[7];
    const float* Wr2 = (const float*)d_in[8];
    const float* Wl3 = (const float*)d_in[9];
    const float* bl3 = (const float*)d_in[10];
    const float* Wr3 = (const float*)d_in[11];
    const float* g1  = (const float*)d_in[12];
    const float* be1 = (const float*)d_in[13];
    const float* g2  = (const float*)d_in[14];
    const float* be2 = (const float*)d_in[15];
    const float* g3  = (const float*)d_in[16];
    const float* be3 = (const float*)d_in[17];
    const float* W_lin1 = (const float*)d_in[18];
    const float* b_lin1 = (const float*)d_in[19];
    const float* W_lin2 = (const float*)d_in[20];
    const float* b_lin2 = (const float*)d_in[21];
    float* out = (float*)d_out;

    // workspace carve-up (256B aligned)
    char* p = (char*)d_ws;
    auto carve = [&](size_t bytes) {
        char* r = p;
        p += (bytes + 255) & ~(size_t)255;
        return r;
    };
    float*     deg   = (float*)carve((size_t)NNODES * 4);
    float*     hA    = (float*)carve((size_t)NNODES * 64 * 4);
    float*     hB    = (float*)carve((size_t)NNODES * 64 * 4);
    float*     accum = (float*)carve((size_t)NNODES * 64 * 4); // also accum6 region
    _Float16*  W16a  = (_Float16*)carve(8192 * 2);
    _Float16*  W16b  = (_Float16*)carve(8192 * 2);
    float*     sums  = (float*)carve(128 * 4);                 // sums[64] + sumsq[64]
    float*     sumsq = sums + 64;
    float*     pmax  = (float*)carve((size_t)NGRAPH * 64 * 4);
    float*     psum  = (float*)carve((size_t)NGRAPH * 64 * 4);
    float*     pcnt  = (float*)carve((size_t)NGRAPH * 4);
    float*     zbuf  = (float*)carve((size_t)NGRAPH * 64 * 4);

    const int NF   = NNODES * 64;
    const int gNF  = (NF + 255) / 256;
    const int gE   = (NEDGES + 255) / 256;
    const int gE16 = (NEDGES * 16 + 255) / 256;
    const int gN   = (NNODES + 255) / 256;
    const int gGemm = (NNODES + 127) / 128;
    const int gStat = (NNODES + 255) / 256;

    // ---- zero accumulation buffers (fresh every call) ----
    hipMemsetAsync(deg,   0, (size_t)NNODES * 4, stream);
    hipMemsetAsync(accum, 0, (size_t)NNODES * 64 * 4, stream);
    hipMemsetAsync(pmax,  0, (size_t)NGRAPH * 64 * 4, stream);
    hipMemsetAsync(psum,  0, (size_t)NGRAPH * 64 * 4, stream);
    hipMemsetAsync(pcnt,  0, (size_t)NGRAPH * 4, stream);

    // ---- layer 1 ----
    deg_scatter6_kernel<<<gE, 256, 0, stream>>>(ei, x, accum, deg, NEDGES);
    invert_deg_kernel<<<gN, 256, 0, stream>>>(deg, NNODES);
    sage1_kernel<<<gNF, 256, 0, stream>>>(x, accum, deg, Wl1, bl1, Wr1, hA, NNODES);
    hipMemsetAsync(sums, 0, 128 * 4, stream);
    colstats_kernel<<<gStat, 256, 0, stream>>>(hA, sums, sumsq, NNODES);
    bn_relu_kernel<<<gNF, 256, 0, stream>>>(hA, sums, sumsq, g1, be1, NNODES);

    // ---- weight prep (f16) ----
    prep_w16_kernel<<<32, 256, 0, stream>>>(Wl2, Wr2, W16a);
    prep_w16_kernel<<<32, 256, 0, stream>>>(Wl3, Wr3, W16b);

    // ---- layer 2 ----
    hipMemsetAsync(accum, 0, (size_t)NNODES * 64 * 4, stream);
    scatter64_kernel<<<gE16, 256, 0, stream>>>(ei, hA, accum, NEDGES);
    sage_gemm_wmma_kernel<<<gGemm, 256, 0, stream>>>(accum, hA, deg, W16a, bl2, hB, NNODES);
    hipMemsetAsync(sums, 0, 128 * 4, stream);
    colstats_kernel<<<gStat, 256, 0, stream>>>(hB, sums, sumsq, NNODES);
    bn_relu_kernel<<<gNF, 256, 0, stream>>>(hB, sums, sumsq, g2, be2, NNODES);

    // ---- layer 3 ----
    hipMemsetAsync(accum, 0, (size_t)NNODES * 64 * 4, stream);
    scatter64_kernel<<<gE16, 256, 0, stream>>>(ei, hB, accum, NEDGES);
    sage_gemm_wmma_kernel<<<gGemm, 256, 0, stream>>>(accum, hB, deg, W16b, bl3, hA, NNODES);
    hipMemsetAsync(sums, 0, 128 * 4, stream);
    colstats_kernel<<<gStat, 256, 0, stream>>>(hA, sums, sumsq, NNODES);
    bn_relu_kernel<<<gNF, 256, 0, stream>>>(hA, sums, sumsq, g3, be3, NNODES);

    // ---- pooling + head ----
    pool_kernel<<<gNF, 256, 0, stream>>>(hA, batch, pmax, psum, pcnt, NNODES);
    mlp1_kernel<<<(NGRAPH * 64 + 255) / 256, 256, 0, stream>>>(pmax, psum, pcnt, W_lin1, b_lin1, zbuf);
    mlp2_kernel<<<(NGRAPH * 2 + 255) / 256, 256, 0, stream>>>(zbuf, W_lin2, b_lin2, out);
}
